// MxDNALearntTokenizationLayer_15513421873292
// MI455X (gfx1250) — compile-verified, matrix operand order checked by reading, not run on hardware
//
#include <hip/hip_runtime.h>
#include <math.h>
#include <stdint.h>

// ---------------- problem constants ----------------
#define BATCH 8
#define SEQ   4096
#define NTOK  (BATCH * SEQ)      // 32768 flat tokens
#define HDIM  768
#define NEXP  6
#define EPS   1e-5f
#define KBLKS (HDIM / 32)        // 24 K-blocks of 32

typedef float  v8f   __attribute__((ext_vector_type(8)));
typedef __bf16 v16bf __attribute__((ext_vector_type(16)));
typedef __bf16 v8bf  __attribute__((ext_vector_type(8)));
typedef int    v2i   __attribute__((ext_vector_type(2)));

__device__ __forceinline__ float sigmoid_f(float x) {
    return __builtin_amdgcn_rcpf(1.0f + __expf(-x));   // v_rcp_f32, not precise div
}

// ---- CDNA5 async global->LDS copy (ASYNCcnt) with safe fallback ----
#if defined(__has_builtin)
#  if __has_builtin(__builtin_amdgcn_global_load_async_to_lds_b64)
#    define HAVE_ASYNC_LDS 1
#  endif
#endif
#ifndef HAVE_ASYNC_LDS
#  define HAVE_ASYNC_LDS 0
#endif

__device__ __forceinline__ void async_copy_b64(const void* g, void* l) {
#if HAVE_ASYNC_LDS
    __builtin_amdgcn_global_load_async_to_lds_b64(
        (__attribute__((address_space(1))) v2i*)(const_cast<void*>(g)),
        (__attribute__((address_space(3))) v2i*)(l),
        0, 0);
#else
    *(double*)l = *(const double*)g;
#endif
}

__device__ __forceinline__ void async_wait0() {
#if HAVE_ASYNC_LDS
#  if __has_builtin(__builtin_amdgcn_s_wait_asynccnt)
    __builtin_amdgcn_s_wait_asynccnt(0);
#  else
    asm volatile("s_wait_asynccnt 0x0" ::: "memory");
#  endif
#endif
}

// ---------------------------------------------------------------------------
// Kernel 0: zero-fill output
// ---------------------------------------------------------------------------
__global__ void zero_out_kernel(float* __restrict__ out, long long n) {
    long long i = (long long)blockIdx.x * blockDim.x + threadIdx.x;
    long long stride = (long long)gridDim.x * blockDim.x;
    for (; i < n; i += stride) out[i] = 0.0f;
}

// ---------------------------------------------------------------------------
// Kernel 1: per-expert ordered center lists (6 blocks x 1024, chunked scan)
// ---------------------------------------------------------------------------
__global__ void build_lists_kernel(const int* __restrict__ mctr,
                                   int* __restrict__ counts,
                                   int* __restrict__ centers) {
    const int e = blockIdx.x;
    const int tid = threadIdx.x;
    __shared__ int s[1024];
    int base = 0;
    for (int c0 = 0; c0 < NTOK; c0 += 1024) {
        const int gpos = c0 + tid;
        const int pred = (mctr[gpos] == e) ? 1 : 0;
        s[tid] = pred;
        __syncthreads();
        for (int off = 1; off < 1024; off <<= 1) {
            int v = (tid >= off) ? s[tid - off] : 0;
            __syncthreads();
            s[tid] += v;
            __syncthreads();
        }
        if (pred) centers[e * NTOK + base + s[tid] - 1] = gpos;
        int tot = s[1023];
        __syncthreads();
        base += tot;
    }
    if (tid == 0) counts[e] = base;
}

// ---------------------------------------------------------------------------
// Kernel 2: pre-swizzle f32 weights -> bf16 WMMA B-fragment layout.
// Fragment element i holds K = koff + (i<8 ? i : i+8), koff = (lane<16?0:8),
// column = nchunk*16 + (lane&15).  GEMM side then loads one 32B vector/lane.
// ---------------------------------------------------------------------------
__global__ void swizzle_w_kernel(const float* __restrict__ W,      // (E, 768, N)
                                 unsigned short* __restrict__ outw,
                                 int N, long long total) {
    __bf16* out = reinterpret_cast<__bf16*>(outw);
    const int nchunks = N >> 4;
    long long idx = (long long)blockIdx.x * blockDim.x + threadIdx.x;
    const long long stride = (long long)gridDim.x * blockDim.x;
    for (; idx < total; idx += stride) {
        const int lane = (int)(idx & 31);
        long long t = idx >> 5;
        const int kblk = (int)(t % KBLKS); t /= KBLKS;
        const int nchunk = (int)(t % nchunks);
        const int e = (int)(t / nchunks);
        const int col = nchunk * 16 + (lane & 15);
        const int koff = (lane < 16) ? 0 : 8;
        const float* src = W + ((size_t)e * HDIM + (size_t)kblk * 32) * N + col;
        __bf16* dst = out + (size_t)idx * 16;
#pragma unroll
        for (int i = 0; i < 16; ++i) {
            const int K = koff + ((i < 8) ? i : (i + 8));
            dst[i] = (__bf16)src[(size_t)K * N];
        }
    }
}

// ---------------------------------------------------------------------------
// Kernel 3: X = GLU(gather(tokens) @ w_pre[e]) -> bf16 (Nt x 768)
// Double-buffered pipeline: async-DMA K-block k+1 into the ping-pong LDS
// buffer while WMMAs consume K-block k; one wait+barrier per K step.
// ---------------------------------------------------------------------------
__global__ void glu_pre_kernel(const float* __restrict__ hidden,
                               const unsigned short* __restrict__ WpreSe, // swizzled, expert base
                               const int*   __restrict__ centers_e,
                               const int*   __restrict__ counts,
                               unsigned short* __restrict__ Xraw,
                               int e, int k) {
    __bf16* X = reinterpret_cast<__bf16*>(Xraw);
    const __bf16* Wb = reinterpret_cast<const __bf16*>(WpreSe);
    __shared__ __align__(16) float Asf[2][16 * 32];

    const int Nc = counts[e];
    const int Nt = Nc * k;
    const int tilesM = (Nt + 15) >> 4;
    const int total = tilesM * 6;
    const int wave = threadIdx.x >> 5;
    const int lane = threadIdx.x & 31;
    const int m    = lane & 15;
    const int koff = (lane < 16) ? 0 : 8;

    // A-stage copy slot for this thread: row = tid>>4, float-pair = tid&15
    const int crow = threadIdx.x >> 4;
    const int cpair = (threadIdx.x & 15) * 2;

    for (int tile = blockIdx.x; tile < total; tile += gridDim.x) {
        const int mt = tile / 6, grp = tile % 6;
        const int mbase = mt << 4;
        const int nA = grp * 8 + wave;          // 0..47 -> GLU cols
        const int nG = nA + 48;                 // gate cols (+768)

        __syncthreads();                        // prev tile readers done
        // zero rows beyond Nt in both buffers (persist across K steps)
        for (int idx = threadIdx.x; idx < 512; idx += 256)
            if (mbase + (idx >> 5) >= Nt) { Asf[0][idx] = 0.0f; Asf[1][idx] = 0.0f; }

        const int trow = mbase + crow;
        const bool rvalid = (trow < Nt);
        const float* gsrc = nullptr;
        if (rvalid) {
            const int pos = centers_e[trow / k] + (trow % k);
            gsrc = hidden + (size_t)pos * HDIM + cpair;
        }

        // pipeline prologue: DMA K-block 0 into buffer 0
        if (rvalid) async_copy_b64(gsrc, &Asf[0][crow * 32 + cpair]);
        async_wait0();
        __syncthreads();

        v8f accA = {}; v8f accG = {};
        for (int kblk = 0; kblk < KBLKS; ++kblk) {
            const int cur = kblk & 1;
            if (rvalid && (kblk + 1 < KBLKS))   // DMA next K-block while computing
                async_copy_b64(gsrc + (kblk + 1) * 32, &Asf[cur ^ 1][crow * 32 + cpair]);

            v16bf a;
#pragma unroll
            for (int i = 0; i < 16; ++i) {
                const int K = koff + ((i < 8) ? i : (i + 8));
                a[i] = (__bf16)Asf[cur][m * 32 + K];
            }
            const __bf16* bp0 = Wb + (((size_t)nA * KBLKS + kblk) * 32 + lane) * 16;
            const __bf16* bp1 = Wb + (((size_t)nG * KBLKS + kblk) * 32 + lane) * 16;
            v16bf b0 = *(const v16bf*)bp0;
            v16bf b1 = *(const v16bf*)bp1;
            __builtin_prefetch(bp0 + 512, 0, 3);   // next kblk fragment, keep near
            __builtin_prefetch(bp1 + 512, 0, 3);

            accA = __builtin_amdgcn_wmma_f32_16x16x32_bf16(false, a, false, b0, (short)0, accA, false, false);
            accG = __builtin_amdgcn_wmma_f32_16x16x32_bf16(false, a, false, b1, (short)0, accG, false, false);

            async_wait0();                      // next buffer landed
            __syncthreads();                    // + everyone done with current
        }

        const int col = nA * 16 + m;
        const int mhi = (lane < 16) ? 0 : 8;
#pragma unroll
        for (int r = 0; r < 8; ++r) {
            const int t = mbase + r + mhi;
            if (t < Nt) {
                const float glu = accA[r] * sigmoid_f(accG[r]);
                X[(size_t)t * HDIM + col] = (__bf16)glu;
            }
        }
    }
}

// ---------------------------------------------------------------------------
// Kernel 4: grouped conv (stride==kernel) + LayerNorm + SiLU per unit.
// Ping-pong: async-DMA the next unit's X tile while computing the current one.
// ---------------------------------------------------------------------------
__global__ void conv_ln_kernel(const unsigned short* __restrict__ Xraw,
                               const float* __restrict__ convw_e,       // (768, gs, k)
                               const float* __restrict__ lnw_e,
                               const float* __restrict__ lnb_e,
                               const int* __restrict__ counts,
                               unsigned short* __restrict__ Yraw,
                               int e, int k, int gs) {
    const __bf16* X = reinterpret_cast<const __bf16*>(Xraw);
    __bf16* Y = reinterpret_cast<__bf16*>(Yraw);
    __shared__ __align__(16) __bf16 xs[2][6 * HDIM];
    __shared__ float rs[256], rq[256];

    const int Nc = counts[e];
    const int tid = threadIdx.x;
    const int ncopy = k * 192;                   // (k*768 bf16) / 4-per-b64
    const int u0 = blockIdx.x;

    if (u0 < Nc)
        for (int idx = tid; idx < ncopy; idx += 256)
            async_copy_b64(X + (size_t)u0 * k * HDIM + idx * 4, &xs[0][idx * 4]);

    int ib = 0;
    for (int u = u0; u < Nc; u += gridDim.x, ib ^= 1) {
        async_wait0();                           // current tile landed
        __syncthreads();
        const int un = u + gridDim.x;
        if (un < Nc)                             // DMA next tile during compute
            for (int idx = tid; idx < ncopy; idx += 256)
                async_copy_b64(X + (size_t)un * k * HDIM + idx * 4, &xs[ib ^ 1][idx * 4]);

        float yv[3];
#pragma unroll
        for (int q = 0; q < 3; ++q) {
            const int o = tid + q * 256;
            const int ibase = (o / gs) * gs;
            const float* w = convw_e + (size_t)o * gs * k;
            float acc = 0.0f;
            for (int c = 0; c < gs; ++c)
                for (int j = 0; j < k; ++j)
                    acc += w[c * k + j] * (float)xs[ib][j * HDIM + ibase + c];
            yv[q] = acc;
        }

        float s  = yv[0] + yv[1] + yv[2];
        float s2 = yv[0]*yv[0] + yv[1]*yv[1] + yv[2]*yv[2];
        rs[tid] = s; rq[tid] = s2;
        __syncthreads();
        for (int off = 128; off > 0; off >>= 1) {
            if (tid < off) { rs[tid] += rs[tid + off]; rq[tid] += rq[tid + off]; }
            __syncthreads();
        }
        const float mu  = rs[0] * (1.0f / HDIM);
        const float var = rq[0] * (1.0f / HDIM) - mu * mu;
        const float inv = rsqrtf(var + EPS);
#pragma unroll
        for (int q = 0; q < 3; ++q) {
            const int o = tid + q * 256;
            float t = (yv[q] - mu) * inv * lnw_e[o] + lnb_e[o];
            t = t * sigmoid_f(t);
            Y[(size_t)u * HDIM + o] = (__bf16)t;
        }
    }
}

// ---------------------------------------------------------------------------
// Kernel 5: out[center] = (Y @ w_post[e]) * softmax(router)[e]
// A (bf16, contiguous rows): two 16B loads/lane, no LDS. B: swizzled bf16.
// ---------------------------------------------------------------------------
__global__ void post_scatter_kernel(const unsigned short* __restrict__ Yraw,
                                    const unsigned short* __restrict__ WpostSe,
                                    const float* __restrict__ router,
                                    const int*   __restrict__ centers_e,
                                    const int*   __restrict__ counts,
                                    float* __restrict__ out, int e) {
    const __bf16* Y  = reinterpret_cast<const __bf16*>(Yraw);
    const __bf16* Wb = reinterpret_cast<const __bf16*>(WpostSe);
    __shared__ float sw[16];
    __shared__ int   ps[16];

    const int Nc = counts[e];
    const int tilesM = (Nc + 15) >> 4;
    const int total = tilesM * 6;
    const int wave = threadIdx.x >> 5;
    const int lane = threadIdx.x & 31;
    const int m    = lane & 15;
    const int koff = (lane < 16) ? 0 : 8;

    for (int tile = blockIdx.x; tile < total; tile += gridDim.x) {
        const int mt = tile / 6, grp = tile % 6;
        const int mbase = mt << 4;
        const int nchunk = grp * 8 + wave;      // 0..47

        __syncthreads();
        if (threadIdx.x < 16) {
            const int row = mbase + threadIdx.x;
            float w = 0.0f; int pos = 0;
            if (row < Nc) {
                pos = centers_e[row];
                const float* rl = router + (size_t)pos * NEXP;
                float mx = rl[0];
#pragma unroll
                for (int i = 1; i < NEXP; ++i) mx = fmaxf(mx, rl[i]);
                float den = 0.0f;
#pragma unroll
                for (int i = 0; i < NEXP; ++i) den += __expf(rl[i] - mx);
                w = __expf(rl[e] - mx) * __builtin_amdgcn_rcpf(den);
            }
            sw[threadIdx.x] = w; ps[threadIdx.x] = pos;
        }
        __syncthreads();

        v8f acc = {};
        const int arow = mbase + m;
        const bool avalid = (arow < Nc);
        const __bf16* yrow = Y + (size_t)arow * HDIM + koff;
        for (int kblk = 0; kblk < KBLKS; ++kblk) {
            v16bf a = {};
            if (avalid) {
                v8bf lo = *(const v8bf*)(yrow + kblk * 32);
                v8bf hi = *(const v8bf*)(yrow + kblk * 32 + 16);
                a = __builtin_shufflevector(lo, hi, 0,1,2,3,4,5,6,7,8,9,10,11,12,13,14,15);
            }
            const __bf16* bp = Wb + (((size_t)nchunk * KBLKS + kblk) * 32 + lane) * 16;
            v16bf b = *(const v16bf*)bp;
            __builtin_prefetch(bp + 512, 0, 3);
            acc = __builtin_amdgcn_wmma_f32_16x16x32_bf16(false, a, false, b, (short)0, acc, false, false);
        }

        const int col = nchunk * 16 + m;
        const int mhi = (lane < 16) ? 0 : 8;
#pragma unroll
        for (int r = 0; r < 8; ++r) {
            const int row = mbase + r + mhi;
            if (row < Nc)
                out[(size_t)ps[r + mhi] * HDIM + col] = acc[r] * sw[r + mhi];
        }
    }
}

// ---------------------------------------------------------------------------
// Host launcher
// ---------------------------------------------------------------------------
extern "C" void kernel_launch(void* const* d_in, const int* in_sizes, int n_in,
                              void* d_out, int out_size, void* d_ws, size_t ws_size,
                              hipStream_t stream) {
    const float* hidden = (const float*)d_in[0];
    const float* router = (const float*)d_in[1];
    const int*   mctr   = (const int*)d_in[2];
    const float* wpre   = (const float*)d_in[4];   // (6,768,1536)
    const float* wpost  = (const float*)d_in[5];   // (6,768,768)
    const float* lnw    = (const float*)d_in[6];
    const float* lnb    = (const float*)d_in[7];
    float* out = (float*)d_out;

    static const int KS[NEXP] = {1, 2, 3, 4, 5, 6};
    static const int GS[NEXP] = {768, 384, 256, 192, 192, 128};

    // ws: counts | centers | X bf16 | Y bf16 | WpreS bf16 | WpostS bf16
    char* ws = (char*)d_ws;
    int* counts  = (int*)ws;
    int* centers = (int*)(ws + 256);
    unsigned short* X      = (unsigned short*)(ws + 256 + (size_t)NEXP * NTOK * 4);
    unsigned short* Ybuf   = (unsigned short*)((char*)X    + (size_t)NTOK * HDIM * 2);
    unsigned short* WpreS  = (unsigned short*)((char*)Ybuf + (size_t)NTOK * HDIM * 2);
    unsigned short* WpostS = (unsigned short*)((char*)WpreS + (size_t)NEXP * HDIM * 2 * HDIM * 2);

    zero_out_kernel<<<4096, 256, 0, stream>>>(out, (long long)NTOK * HDIM);
    build_lists_kernel<<<NEXP, 1024, 0, stream>>>(mctr, counts, centers);
    swizzle_w_kernel<<<1024, 256, 0, stream>>>(wpre,  WpreS,  2 * HDIM,
                                               (long long)NEXP * 96 * KBLKS * 32);
    swizzle_w_kernel<<<1024, 256, 0, stream>>>(wpost, WpostS, HDIM,
                                               (long long)NEXP * 48 * KBLKS * 32);

    const size_t preFragPerExp  = (size_t)96 * KBLKS * 32 * 16;  // bf16 elems
    const size_t postFragPerExp = (size_t)48 * KBLKS * 32 * 16;

    for (int e = 0; e < NEXP; ++e) {
        const int k  = KS[e];
        const int gs = GS[e];
        const float* convw_e = (const float*)d_in[8 + e];
        const int*   cent_e  = centers + (size_t)e * NTOK;

        glu_pre_kernel<<<2048, 256, 0, stream>>>(hidden, WpreS + e * preFragPerExp,
                                                 cent_e, counts, X, e, k);
        conv_ln_kernel<<<2048, 256, 0, stream>>>(X, convw_e, lnw + (size_t)e * HDIM,
                                                 lnb + (size_t)e * HDIM, counts, Ybuf, e, k, gs);
        post_scatter_kernel<<<1024, 256, 0, stream>>>(Ybuf, WpostS + e * postFragPerExp,
                                                      router, cent_e, counts, out, e);
    }
}
